// Decoder_20925080666825
// MI455X (gfx1250) — compile-verified, matrix operand order checked by reading, not run on hardware
//
#include <hip/hip_runtime.h>
#include <hip/hip_bf16.h>
#include <math.h>

// ---- problem constants (match reference) ----
#define Bsz 32
#define Ssz 100
#define Tsz 64
#define Hsz 512
#define Esz 512
#define Vsz 32000

typedef __attribute__((ext_vector_type(16))) __bf16 v16bf;
typedef __attribute__((ext_vector_type(8)))  __bf16 v8bf;
typedef __attribute__((ext_vector_type(8)))  float  v8f;

// ---------------------------------------------------------------------------
// WMMA helper: D = A(16x32 bf16) x B(32x16 bf16) + C(16x16 f32)
// A layout (ISA 7.12.2, 16-bit A 16x32): lane L holds row m = L%16;
//   kb = 8*(L/16); frag elems [0..7] = K=kb..kb+7, [8..15] = K=kb+16..kb+23.
// B mirrored with lane -> column. C/D: VGPR r, lane L -> row r+8*(L/16), col L%16.
// ---------------------------------------------------------------------------
__device__ inline v8f wmma_bf16(v16bf a, v16bf b, v8f c) {
  return __builtin_amdgcn_wmma_f32_16x16x32_bf16(
      /*neg_a=*/false, a, /*neg_b=*/false, b,
      /*c_mod=*/(short)0, c, /*reuse_a=*/false, /*reuse_b=*/false);
}

// --- fragment loaders (lane base pre-offset by kb; 16B-aligned by layout) ---
// fp32 source: four b128 loads + v_cvt_pk_bf16_f32 (fallback path).
__device__ inline v16bf ldfrag(const float* __restrict__ p) {
  float4 a0 = *(const float4*)p,        a1 = *(const float4*)(p + 4);
  float4 a2 = *(const float4*)(p + 16), a3 = *(const float4*)(p + 20);
  v16bf r;
  r[0] = (__bf16)a0.x; r[1] = (__bf16)a0.y; r[2]  = (__bf16)a0.z; r[3]  = (__bf16)a0.w;
  r[4] = (__bf16)a1.x; r[5] = (__bf16)a1.y; r[6]  = (__bf16)a1.z; r[7]  = (__bf16)a1.w;
  r[8] = (__bf16)a2.x; r[9] = (__bf16)a2.y; r[10] = (__bf16)a2.z; r[11] = (__bf16)a2.w;
  r[12] = (__bf16)a3.x; r[13] = (__bf16)a3.y; r[14] = (__bf16)a3.z; r[15] = (__bf16)a3.w;
  return r;
}
// bf16 source: exactly two global_load_b128, zero VALU (fast path).
__device__ inline v16bf ldfrag(const __bf16* __restrict__ p) {
  union { uint4 u; v8bf v; } lo, hi;
  lo.u = *(const uint4*)p;
  hi.u = *(const uint4*)(p + 16);
  v16bf r;
#pragma unroll
  for (int j = 0; j < 8; ++j) { r[j] = lo.v[j]; r[8 + j] = hi.v[j]; }
  return r;
}
// fp32 source with K strided by ld floats (setup GEMM B operand).
__device__ inline v16bf ldfrag_s(const float* __restrict__ p, int ld) {
  v16bf r;
#pragma unroll
  for (int j = 0; j < 8; ++j) r[j] = (__bf16)p[j * ld];
#pragma unroll
  for (int j = 0; j < 8; ++j) r[8 + j] = (__bf16)p[(16 + j) * ld];
  return r;
}

__device__ inline float sigm(float x) { return 1.f / (1.f + __expf(-x)); }

// ---------------------------------------------------------------------------
// fp32 -> bf16 array conversion (per-launch weight pre-conversion). n % 1024 == 0.
// ---------------------------------------------------------------------------
__global__ __launch_bounds__(256) void k_cvt_bf16(
    const float* __restrict__ s, __bf16* __restrict__ d, long n) {
  long i = (long)blockIdx.x * 1024 + (long)threadIdx.x * 4;
  float4 v = *(const float4*)(s + i);
  union { __bf16 h[4]; uint2 u; } o;
  o.h[0] = (__bf16)v.x; o.h[1] = (__bf16)v.y;
  o.h[2] = (__bf16)v.z; o.h[3] = (__bf16)v.w;
  *(uint2*)(d + i) = o.u;
}

// ---------------------------------------------------------------------------
// Setup 1 (once per launch): Wqb = Wb @ Wq  (512x512, K=512), output f32 or bf16.
// ---------------------------------------------------------------------------
template <typename OT>
__global__ __launch_bounds__(256) void k_setup_wqb(
    const float* __restrict__ Wb, const float* __restrict__ Wq,
    OT* __restrict__ Wqb) {
  int tid = threadIdx.x, wid = tid >> 5, L = tid & 31;
  int tile = blockIdx.x * 8 + wid;
  int ti = tile >> 5, tj = tile & 31;
  int kb = (L >> 4) * 8;
  const float* a = Wb + (size_t)(ti * 16 + (L & 15)) * Hsz + kb;     // K contig
  const float* b = Wq + (size_t)kb * Hsz + (tj * 16 + (L & 15));     // K strided
  v8f acc = {};
  for (int kc = 0; kc < Hsz; kc += 32)
    acc = wmma_bf16(ldfrag(a + kc), ldfrag_s(b + (size_t)kc * Hsz, Hsz), acc);
#pragma unroll
  for (int r = 0; r < 8; ++r) {
    int row = ti * 16 + r + 8 * (L >> 4);
    Wqb[(size_t)row * Hsz + tj * 16 + (L & 15)] = (OT)acc[r];
  }
}

// ---------------------------------------------------------------------------
// Setup 2: bqb = Wb@bq + bb, copy init state (+ optional bf16 copies).
// grid 194: blocks 0-1 -> bqb, blocks 2..193 -> copies.
// ---------------------------------------------------------------------------
__global__ __launch_bounds__(256) void k_setup_misc(
    const float* __restrict__ Wb, const float* __restrict__ bq,
    const float* __restrict__ bb, const float* __restrict__ ih,
    const float* __restrict__ ic, const float* __restrict__ io,
    float* __restrict__ bqb, float* __restrict__ h0,
    float* __restrict__ cb, float* __restrict__ o0,
    __bf16* __restrict__ h0bf, __bf16* __restrict__ o0bf) {
  int blk = blockIdx.x, tid = threadIdx.x;
  if (blk < 2) {
    int i = blk * 256 + tid;
    float s = bb[i];
    for (int j = 0; j < Hsz; ++j) s += Wb[(size_t)i * Hsz + j] * bq[j];
    bqb[i] = s;
  } else {
    int e = (blk - 2) * 256 + tid;
    if (e < Bsz * Hsz) {
      h0[e] = ih[e];
      if (h0bf) h0bf[e] = (__bf16)ih[e];
    } else if (e < 2 * Bsz * Hsz) {
      cb[e - Bsz * Hsz] = ic[e - Bsz * Hsz];
    } else {
      int i = e - 2 * Bsz * Hsz;
      o0[i] = io[i];
      if (o0bf) o0bf[i] = (__bf16)io[i];
    }
  }
}

// ---------------------------------------------------------------------------
// K1: gates = [emb[trg[:,t]], out_prev] @ W_ih^T + h_prev @ W_hh^T + biases;
//     LSTM pointwise -> h_new (f32 + bf16), c (in-place).
// grid 32; 8 waves = 4 gates x 2 row halves.
// AT: type of activations/emb read (bf16 fast / f32 fallback).
// ---------------------------------------------------------------------------
template <typename AT, typename WT>
__global__ __launch_bounds__(256) void k_gates(
    const AT* __restrict__ emb, const int* __restrict__ trg,
    const WT* __restrict__ W_ih, const WT* __restrict__ W_hh,
    const float* __restrict__ b_ih, const float* __restrict__ b_hh,
    const AT* __restrict__ h_r, const AT* __restrict__ o_r,
    float* __restrict__ cbuf, float* __restrict__ h_w,
    __bf16* __restrict__ h_w_bf, int t) {
  __shared__ float lds[4 * Bsz * 16];
  int tid = threadIdx.x, wid = tid >> 5, L = tid & 31;
  int g = wid & 3, rh = wid >> 2;
  int m = rh * 16 + (L & 15);
  int kb = (L >> 4) * 8;
  int token = trg[m * Tsz + t];
  const AT* aE = emb + (size_t)token * Esz + kb;
  const AT* aO = o_r + (size_t)m * Esz + kb;
  const AT* aH = h_r + (size_t)m * Hsz + kb;
  int nW = g * Hsz + blockIdx.x * 16 + (L & 15);
  const WT* bI = W_ih + (size_t)nW * (2 * Esz) + kb;
  const WT* bH = W_hh + (size_t)nW * Hsz + kb;

  v8f acc = {};
  for (int kc = 0; kc < Esz; kc += 32)
    acc = wmma_bf16(ldfrag(aE + kc), ldfrag(bI + kc), acc);
  for (int kc = 0; kc < Esz; kc += 32)
    acc = wmma_bf16(ldfrag(aO + kc), ldfrag(bI + Esz + kc), acc);
  for (int kc = 0; kc < Hsz; kc += 32)
    acc = wmma_bf16(ldfrag(aH + kc), ldfrag(bH + kc), acc);

  float bias = b_ih[nW] + b_hh[nW];
#pragma unroll
  for (int r = 0; r < 8; ++r) {
    int mc = rh * 16 + r + 8 * (L >> 4);
    lds[g * (Bsz * 16) + mc * 16 + (L & 15)] = acc[r] + bias;
  }
  __syncthreads();

  for (int e = tid; e < Bsz * 16; e += 256) {
    int b = e >> 4, j = e & 15;
    int col = blockIdx.x * 16 + j;
    float ig = lds[0 * (Bsz * 16) + e];
    float fg = lds[1 * (Bsz * 16) + e];
    float gg = lds[2 * (Bsz * 16) + e];
    float og = lds[3 * (Bsz * 16) + e];
    float cn = sigm(fg) * cbuf[b * Hsz + col] + sigm(ig) * tanhf(gg);
    float hn = sigm(og) * tanhf(cn);
    cbuf[b * Hsz + col] = cn;
    h_w[b * Hsz + col] = hn;
    if (h_w_bf) h_w_bf[b * Hsz + col] = (__bf16)hn;
  }
}

// ---------------------------------------------------------------------------
// Generic C(32xN) = A(32xK) @ W^T(NxK) + bias, optional tanh + bf16 mirror.
// grid = N/64.
// ---------------------------------------------------------------------------
template <int ACT, typename AT, typename WT>
__global__ __launch_bounds__(256) void k_gemm32(
    const AT* __restrict__ A, int K, const WT* __restrict__ W,
    const float* __restrict__ bias, float* __restrict__ C, long ldc,
    __bf16* __restrict__ Cbf) {
  int tid = threadIdx.x, wid = tid >> 5, L = tid & 31;
  int n0 = blockIdx.x * 64 + (wid & 3) * 16;
  int m0 = (wid >> 2) * 16;
  int kb = (L >> 4) * 8;
  const AT* a = A + (size_t)(m0 + (L & 15)) * K + kb;
  const WT* w = W + (size_t)(n0 + (L & 15)) * K + kb;
  v8f acc = {};
  for (int kc = 0; kc < K; kc += 32)
    acc = wmma_bf16(ldfrag(a + kc), ldfrag(w + kc), acc);
  float bv = bias ? bias[n0 + (L & 15)] : 0.f;
#pragma unroll
  for (int r = 0; r < 8; ++r) {
    int row = m0 + r + 8 * (L >> 4);
    float v = acc[r] + bv;
    if (ACT == 1) v = tanhf(v);
    C[(size_t)row * ldc + n0 + (L & 15)] = v;
    if (Cbf) Cbf[(size_t)row * ldc + n0 + (L & 15)] = (__bf16)v;
  }
}

// ---------------------------------------------------------------------------
// K4: tanh([h, ctx] @ Wo1^T + bo1)  (A = concat), f32 + bf16 outputs.
// ---------------------------------------------------------------------------
template <typename AT, typename WT>
__global__ __launch_bounds__(256) void k_gemm32_cat(
    const AT* __restrict__ A1, const AT* __restrict__ A2,
    const WT* __restrict__ W, const float* __restrict__ bias,
    float* __restrict__ C, __bf16* __restrict__ Cbf) {
  int tid = threadIdx.x, wid = tid >> 5, L = tid & 31;
  int n0 = blockIdx.x * 64 + (wid & 3) * 16;
  int m0 = (wid >> 2) * 16;
  int kb = (L >> 4) * 8;
  const AT* a1 = A1 + (size_t)(m0 + (L & 15)) * Hsz + kb;
  const AT* a2 = A2 + (size_t)(m0 + (L & 15)) * Hsz + kb;
  const WT* w = W + (size_t)(n0 + (L & 15)) * (2 * Hsz) + kb;
  v8f acc = {};
  for (int kc = 0; kc < Hsz; kc += 32)
    acc = wmma_bf16(ldfrag(a1 + kc), ldfrag(w + kc), acc);
  for (int kc = 0; kc < Hsz; kc += 32)
    acc = wmma_bf16(ldfrag(a2 + kc), ldfrag(w + Hsz + kc), acc);
  float bv = bias[n0 + (L & 15)];
#pragma unroll
  for (int r = 0; r < 8; ++r) {
    int row = m0 + r + 8 * (L >> 4);
    float v = tanhf(acc[r] + bv);
    C[(size_t)row * Hsz + n0 + (L & 15)] = v;
    if (Cbf) Cbf[(size_t)row * Hsz + n0 + (L & 15)] = (__bf16)v;
  }
}

// ---------------------------------------------------------------------------
// K6: logits (32 x 32000) = out @ emb^T. Dominant GEMM.
// Each wave: 1 m-tile x 4 n-tiles, A-fragment amortized over 4 WMMAs/chunk.
// WG covers 256 columns -> grid 125.
// ---------------------------------------------------------------------------
template <typename AT, typename WT>
__global__ __launch_bounds__(256) void k_logits(
    const AT* __restrict__ A, const WT* __restrict__ W,
    float* __restrict__ C, long ldc) {
  int tid = threadIdx.x, wid = tid >> 5, L = tid & 31;
  int m0 = (wid & 1) * 16;
  int n0 = blockIdx.x * 256 + (wid >> 1) * 64;
  int kb = (L >> 4) * 8;
  const AT* a = A + (size_t)(m0 + (L & 15)) * Esz + kb;
  const WT* w0 = W + (size_t)(n0 + 0  + (L & 15)) * Esz + kb;
  const WT* w1 = W + (size_t)(n0 + 16 + (L & 15)) * Esz + kb;
  const WT* w2 = W + (size_t)(n0 + 32 + (L & 15)) * Esz + kb;
  const WT* w3 = W + (size_t)(n0 + 48 + (L & 15)) * Esz + kb;
  v8f acc0 = {}, acc1 = {}, acc2 = {}, acc3 = {};
  for (int kc = 0; kc < Esz; kc += 32) {
    v16bf av = ldfrag(a + kc);
    acc0 = wmma_bf16(av, ldfrag(w0 + kc), acc0);
    acc1 = wmma_bf16(av, ldfrag(w1 + kc), acc1);
    acc2 = wmma_bf16(av, ldfrag(w2 + kc), acc2);
    acc3 = wmma_bf16(av, ldfrag(w3 + kc), acc3);
  }
#pragma unroll
  for (int r = 0; r < 8; ++r) {
    int row = m0 + r + 8 * (L >> 4);
    float* cr = C + (size_t)row * ldc + n0 + (L & 15);
    cr[0] = acc0[r]; cr[16] = acc1[r]; cr[32] = acc2[r]; cr[48] = acc3[r];
  }
}

// ---------------------------------------------------------------------------
// K3: attention. One WG per batch row; ctx written f32 (+ optional bf16).
// ---------------------------------------------------------------------------
__global__ __launch_bounds__(256) void k_attn(
    const float* __restrict__ qb, const float* __restrict__ srcmem,
    const unsigned char* __restrict__ mask, float* __restrict__ ctx,
    __bf16* __restrict__ ctxbf) {
  __shared__ float q_s[Hsz];
  __shared__ float sc[128];
  __shared__ float invsum;
  int b = blockIdx.x, tid = threadIdx.x;
  const float* src = srcmem + (size_t)b * Ssz * Hsz;
  q_s[tid] = qb[b * Hsz + tid];
  q_s[tid + 256] = qb[b * Hsz + tid + 256];
  __syncthreads();
  int wid = tid >> 5, L = tid & 31;
  for (int s = wid; s < Ssz; s += 8) {
    const float* row = src + (size_t)s * Hsz;
    float p = 0.f;
    for (int k = L; k < Hsz; k += 32) p += q_s[k] * row[k];
#pragma unroll
    for (int off = 16; off; off >>= 1) p += __shfl_xor(p, off, 32);
    if (L == 0) sc[s] = mask[b * Ssz + s] ? p : -1e9f;
  }
  __syncthreads();
  if (tid == 0) {
    float mx = -3.4e38f;
    for (int s = 0; s < Ssz; ++s) mx = fmaxf(mx, sc[s]);
    float sum = 0.f;
    for (int s = 0; s < Ssz; ++s) { float e = __expf(sc[s] - mx); sc[s] = e; sum += e; }
    invsum = 1.f / sum;
  }
  __syncthreads();
  float inv = invsum;
  for (int col = tid; col < Hsz; col += 256) {
    float acc = 0.f;
    for (int s = 0; s < Ssz; ++s) acc += sc[s] * src[(size_t)s * Hsz + col];
    float v = acc * inv;
    ctx[b * Hsz + col] = v;
    if (ctxbf) ctxbf[b * Hsz + col] = (__bf16)v;
  }
}

// ---------------------------------------------------------------------------
// Orchestration: per-launch setup + 64 steps, all on `stream` (capture-safe).
// Fast path (workspace permitting): every WMMA operand pre-converted to bf16
// (weights once per launch, activations at their producer) -> hot GEMM loops
// are pure global_load_b128 + v_wmma. Fallback: fp32 sources, convert inline.
// ---------------------------------------------------------------------------
extern "C" void kernel_launch(void* const* d_in, const int* in_sizes, int n_in,
                              void* d_out, int out_size, void* d_ws, size_t ws_size,
                              hipStream_t stream) {
  const float* srcmem = (const float*)d_in[0];
  const unsigned char* mask = (const unsigned char*)d_in[1];
  const float* ih   = (const float*)d_in[2];
  const float* ic   = (const float*)d_in[3];
  const float* io   = (const float*)d_in[4];
  const int*   trg  = (const int*)d_in[5];
  const float* emb  = (const float*)d_in[6];
  const float* W_ih = (const float*)d_in[7];
  const float* W_hh = (const float*)d_in[8];
  const float* b_ih = (const float*)d_in[9];
  const float* b_hh = (const float*)d_in[10];
  const float* Wq   = (const float*)d_in[11];
  const float* bq   = (const float*)d_in[12];
  const float* Wb   = (const float*)d_in[13];
  const float* bb   = (const float*)d_in[14];
  const float* Wo1  = (const float*)d_in[15];
  const float* bo1  = (const float*)d_in[16];
  const float* Wo2  = (const float*)d_in[17];
  const float* bo2  = (const float*)d_in[18];
  float* out = (float*)d_out;

  const int BH = Bsz * Hsz;  // 16384
  // f32 region (both paths)
  float* ws  = (float*)d_ws;
  float* h0  = ws;
  float* h1  = h0 + BH;
  float* cb  = h1 + BH;
  float* o0  = cb + BH;
  float* o1  = o0 + BH;
  float* qb  = o1 + BH;
  float* ctx = qb + BH;
  float* tv  = ctx + BH;
  float* bqb = tv + BH;              // 512
  float* WqbF = bqb + Hsz;           // 512*512 f32 (fallback path)
  size_t f32_bytes = (size_t)(8 * BH + Hsz + Hsz * Hsz) * 4;   // 1,574,912 B

  // bf16 region (fast path): weights + activation mirrors
  __bf16* bfbase = (__bf16*)((char*)d_ws + f32_bytes);
  __bf16* WqbB = bfbase;                        // 512*512
  __bf16* embB = WqbB + Hsz * Hsz;              // 32000*512
  __bf16* WihB = embB + (size_t)Vsz * Esz;      // 2048*1024
  __bf16* WhhB = WihB + (size_t)4 * Hsz * 2 * Esz;
  __bf16* Wo1B = WhhB + (size_t)4 * Hsz * Hsz;  // 512*1024
  __bf16* Wo2B = Wo1B + (size_t)Hsz * 2 * Hsz;  // 512*512
  __bf16* hbf0 = Wo2B + (size_t)Hsz * Hsz;      // activation mirrors (6 x BH)
  __bf16* hbf1 = hbf0 + BH;
  __bf16* obf0 = hbf1 + BH;
  __bf16* obf1 = obf0 + BH;
  __bf16* ctxB = obf1 + BH;
  __bf16* tvB  = ctxB + BH;
  size_t bf16_elems = (size_t)Hsz * Hsz + (size_t)Vsz * Esz +
                      (size_t)4 * Hsz * 2 * Esz + (size_t)4 * Hsz * Hsz +
                      (size_t)Hsz * 2 * Hsz + (size_t)Hsz * Hsz +
                      (size_t)6 * BH;
  bool fast = ws_size >= f32_bytes + bf16_elems * 2;

  float* hb[2] = {h0, h1};
  float* ob[2] = {o0, o1};
  __bf16* hbf[2] = {hbf0, hbf1};
  __bf16* obf[2] = {obf0, obf1};

  if (fast) {
    k_setup_misc<<<194, 256, 0, stream>>>(Wb, bq, bb, ih, ic, io, bqb, h0, cb, o0,
                                          hbf0, obf0);
    k_setup_wqb<__bf16><<<128, 256, 0, stream>>>(Wb, Wq, WqbB);
    k_cvt_bf16<<<(Vsz * Esz) / 1024, 256, 0, stream>>>(emb, embB, (long)Vsz * Esz);
    k_cvt_bf16<<<(4 * Hsz * 2 * Esz) / 1024, 256, 0, stream>>>(W_ih, WihB, 4L * Hsz * 2 * Esz);
    k_cvt_bf16<<<(4 * Hsz * Hsz) / 1024, 256, 0, stream>>>(W_hh, WhhB, 4L * Hsz * Hsz);
    k_cvt_bf16<<<(Hsz * 2 * Hsz) / 1024, 256, 0, stream>>>(Wo1, Wo1B, (long)Hsz * 2 * Hsz);
    k_cvt_bf16<<<(Hsz * Hsz) / 1024, 256, 0, stream>>>(Wo2, Wo2B, (long)Hsz * Hsz);
    for (int t = 0; t < Tsz; ++t) {
      int r = t & 1, w = r ^ 1;
      k_gates<__bf16, __bf16><<<32, 256, 0, stream>>>(
          embB, trg, WihB, WhhB, b_ih, b_hh, hbf[r], obf[r], cb, hb[w], hbf[w], t);
      k_gemm32<0, __bf16, __bf16><<<8, 256, 0, stream>>>(
          hbf[w], Hsz, WqbB, bqb, qb, Hsz, (__bf16*)nullptr);
      k_attn<<<Bsz, 256, 0, stream>>>(qb, srcmem, mask, ctx, ctxB);
      k_gemm32_cat<__bf16, __bf16><<<8, 256, 0, stream>>>(
          hbf[w], ctxB, Wo1B, bo1, tv, tvB);
      k_gemm32<0, __bf16, __bf16><<<8, 256, 0, stream>>>(
          tvB, Hsz, Wo2B, bo2, ob[w], Hsz, obf[w]);
      k_logits<__bf16, __bf16><<<Vsz / 256, 256, 0, stream>>>(
          obf[w], embB, out + (size_t)t * Vsz, (long)Tsz * Vsz);
    }
  } else {
    k_setup_misc<<<194, 256, 0, stream>>>(Wb, bq, bb, ih, ic, io, bqb, h0, cb, o0,
                                          (__bf16*)nullptr, (__bf16*)nullptr);
    k_setup_wqb<float><<<128, 256, 0, stream>>>(Wb, Wq, WqbF);
    for (int t = 0; t < Tsz; ++t) {
      int r = t & 1, w = r ^ 1;
      k_gates<float, float><<<32, 256, 0, stream>>>(
          emb, trg, W_ih, W_hh, b_ih, b_hh, hb[r], ob[r], cb, hb[w],
          (__bf16*)nullptr, t);
      k_gemm32<0, float, float><<<8, 256, 0, stream>>>(
          hb[w], Hsz, WqbF, bqb, qb, Hsz, (__bf16*)nullptr);
      k_attn<<<Bsz, 256, 0, stream>>>(qb, srcmem, mask, ctx, (__bf16*)nullptr);
      k_gemm32_cat<float, float><<<8, 256, 0, stream>>>(
          hb[w], ctx, Wo1, bo1, tv, (__bf16*)nullptr);
      k_gemm32<0, float, float><<<8, 256, 0, stream>>>(
          tv, Hsz, Wo2, bo2, ob[w], Hsz, (__bf16*)nullptr);
      k_logits<float, float><<<Vsz / 256, 256, 0, stream>>>(
          ob[w], emb, out + (size_t)t * Vsz, (long)Tsz * Vsz);
    }
  }
}
